// MultiHeadAttentionLayer_532575945124
// MI455X (gfx1250) — matrix operand order repaired and checked by reading, not compiled, not run on hardware
//
#include <hip/hip_runtime.h>

#define D_MODEL  1024
#define S_LEN    4096
#define N_HEADS  16
#define HEAD_DIM 64

typedef __attribute__((ext_vector_type(16))) _Float16 v16h;
typedef __attribute__((ext_vector_type(8)))  _Float16 v8h;
typedef __attribute__((ext_vector_type(4)))  _Float16 v4h;
typedef __attribute__((ext_vector_type(8)))  float    v8f;
typedef __attribute__((ext_vector_type(4)))  float    v4f;

// D = A(16x32 f16) * B(32x16 f16) + C(16x16 f32)
__device__ __forceinline__ v8f wmma16(v16h a, v16h b, v8f c) {
  return __builtin_amdgcn_wmma_f32_16x16x32_f16(false, a, false, b, (short)0, c,
                                                false, false);
}

// A-fragment (16x32, 16-bit): lane half picks kbase (0 or 8); elems 0..7 at
// row[kb..kb+8), elems 8..15 at row[kb+16..kb+24).
__device__ __forceinline__ v16h frag_a(const _Float16* __restrict__ row, int kb) {
  v8h lo = *(const v8h*)(row + kb);
  v8h hi = *(const v8h*)(row + kb + 16);
  v16h r;
#pragma unroll
  for (int i = 0; i < 8; ++i) { r[i] = lo[i]; r[i + 8] = hi[i]; }
  return r;
}

// B-fragment (32x16, 16-bit): elems 0..7 at p[0..8), elems 8..15 at p[16..24).
__device__ __forceinline__ v16h frag_b(const _Float16* __restrict__ p) {
  v8h lo = *(const v8h*)(p);
  v8h hi = *(const v8h*)(p + 16);
  v16h r;
#pragma unroll
  for (int i = 0; i < 8; ++i) { r[i] = lo[i]; r[i + 8] = hi[i]; }
  return r;
}

// ---------------------------------------------------------------------------
// Weight transpose + f32 -> f16 convert: WT[n][k] = (f16)W[k][n]
// ---------------------------------------------------------------------------
__global__ __launch_bounds__(256) void wtrans_f16(const float* __restrict__ W,
                                                  _Float16* __restrict__ WT) {
  __shared__ float tile[32][33];
  const int bx = blockIdx.x * 32, by = blockIdx.y * 32;
  const int tx = threadIdx.x & 31, ty = threadIdx.x >> 5;  // 32 x 8
#pragma unroll
  for (int i = ty; i < 32; i += 8)
    tile[i][tx] = W[(size_t)(by + i) * D_MODEL + bx + tx];
  __syncthreads();
#pragma unroll
  for (int i = ty; i < 32; i += 8)
    WT[(size_t)(bx + i) * D_MODEL + by + tx] = (_Float16)tile[tx][i];
}

// ---------------------------------------------------------------------------
// Projection GEMM: Y = (X f32 @ W + b) * scale, written as f16.
// MODE 0: per-head row-major   out[h][s][d]     (Q, K)
// MODE 1: per-head transposed  out[h][d][s]     (V^T)
// Block: 256 threads (8 waves). Tile M=128, N=128 (8 n-tiles/wave), Kstep=32.
// ---------------------------------------------------------------------------
template <int MODE>
__global__ __launch_bounds__(256) void proj_gemm(
    const float* __restrict__ X, const _Float16* __restrict__ WT,
    const float* __restrict__ bias, _Float16* __restrict__ out, float scale) {
  __shared__ _Float16 Xs[128][32];
  __shared__ _Float16 Ws[128][32];
  const int t = threadIdx.x;
  const int lane = t & 31, wave = t >> 5;
  const int m0 = blockIdx.x * 128;
  const int n0 = blockIdx.y * 128;
  const int col = lane & 15;
  const int rbase = (lane >> 4) * 8;
  const int kbA = (lane >> 4) * 8;
  const int kbB = (lane >> 4) * 8;

  v8f acc[8] = {};

  for (int k0 = 0; k0 < D_MODEL; k0 += 32) {
    __syncthreads();
    // stage X tile (128x32 f32 -> f16): 1024 groups of 4, 4 per thread
#pragma unroll
    for (int it = 0; it < 4; ++it) {
      int g = t + it * 256;
      int row = g >> 3;
      int cg = (g & 7) * 4;
      v4f v = *(const v4f*)(X + (size_t)(m0 + row) * D_MODEL + k0 + cg);
      v4h h = {(_Float16)v.x, (_Float16)v.y, (_Float16)v.z, (_Float16)v.w};
      *(v4h*)&Xs[row][cg] = h;
    }
    // stage W^T tile (128x32 f16): 512 groups of 8, 2 per thread
#pragma unroll
    for (int it = 0; it < 2; ++it) {
      int g = t + it * 256;
      int row = g >> 2;
      int cg = (g & 3) * 8;
      *(v8h*)&Ws[row][cg] =
          *(const v8h*)(WT + (size_t)(n0 + row) * D_MODEL + k0 + cg);
    }
    __syncthreads();

    v16h a = frag_a(&Xs[wave * 16 + col][0], kbA);
#pragma unroll
    for (int nt = 0; nt < 8; ++nt) {
      v16h b = frag_b(&Ws[nt * 16 + col][kbB]);
      acc[nt] = wmma16(a, b, acc[nt]);
    }
  }

#pragma unroll
  for (int nt = 0; nt < 8; ++nt) {
    int n = n0 + nt * 16 + col;
    float bv = bias[n];
    int h = n >> 6, d = n & 63;
#pragma unroll
    for (int i = 0; i < 8; ++i) {
      int m = m0 + wave * 16 + rbase + i;
      float val = (acc[nt][i] + bv) * scale;
      if (MODE == 0)
        out[((size_t)h * S_LEN + m) * HEAD_DIM + d] = (_Float16)val;
      else
        out[((size_t)h * HEAD_DIM + d) * S_LEN + m] = (_Float16)val;
    }
  }
}

// ---------------------------------------------------------------------------
// Flash attention: block = (1 head, 128 q rows); 8 waves x 16 rows each.
// Q pre-scaled by 1/sqrt(64). K tile [kv][d], V^T tile [d][kv] staged in LDS.
// ---------------------------------------------------------------------------
__global__ __launch_bounds__(256) void flash_attn(
    const _Float16* __restrict__ qb, const _Float16* __restrict__ kbuf,
    const _Float16* __restrict__ vtb, const int* __restrict__ amask,
    _Float16* __restrict__ attn) {
  __shared__ _Float16 Ks[64][64];       // [kv][d]
  __shared__ _Float16 Vs[64][64];       // [d][kv]
  __shared__ _Float16 Pl[8][16][64];    // per-wave P staging
  const int t = threadIdx.x;
  const int lane = t & 31, wave = t >> 5;
  const int head = blockIdx.y;
  const int q0 = blockIdx.x * 128;
  const int col = lane & 15;
  const int rbase = (lane >> 4) * 8;
  const int kbA = (lane >> 4) * 8;
  const int kbB = (lane >> 4) * 8;

  const _Float16* qhead = qb + (size_t)head * S_LEN * HEAD_DIM;
  const _Float16* khead = kbuf + (size_t)head * S_LEN * HEAD_DIM;
  const _Float16* vhead = vtb + (size_t)head * HEAD_DIM * S_LEN;

  // Q fragments for the wave's 16 rows, both 32-wide k-steps of head_dim=64
  const int arow = q0 + wave * 16 + col;
  v16h aq[2];
#pragma unroll
  for (int s = 0; s < 2; ++s)
    aq[s] = frag_a(qhead + (size_t)arow * HEAD_DIM, kbA + s * 32);

  v8f acc[4] = {};
  float mrow[8], lrow[8];
#pragma unroll
  for (int i = 0; i < 8; ++i) { mrow[i] = -3.0e38f; lrow[i] = 0.0f; }

  for (int kv0 = 0; kv0 < S_LEN; kv0 += 64) {
    __syncthreads();
    // stage K (64x64) and V^T (64x64): 512 groups of 8 each, 2/thread
#pragma unroll
    for (int it = 0; it < 2; ++it) {
      int g = t + it * 256;
      int row = g >> 3;
      int cg = (g & 7) * 8;
      *(v8h*)&Ks[row][cg] =
          *(const v8h*)(khead + (size_t)(kv0 + row) * HEAD_DIM + cg);
      *(v8h*)&Vs[row][cg] =
          *(const v8h*)(vhead + (size_t)row * S_LEN + kv0 + cg);
    }
    __syncthreads();

    // scores S = Qhat @ K^T  (16 x 64), B-frag = contiguous K rows
    v8f sc[4];
#pragma unroll
    for (int nt = 0; nt < 4; ++nt) {
      v8f s0 = {};
#pragma unroll
      for (int s = 0; s < 2; ++s)
        s0 = wmma16(aq[s], frag_b(&Ks[nt * 16 + col][kbB + s * 32]), s0);
      if (amask[kv0 + nt * 16 + col] == 0) {
#pragma unroll
        for (int i = 0; i < 8; ++i) s0[i] = -1.0e9f;
      }
      sc[nt] = s0;
    }

    // online softmax: rows live across 16-lane halves -> xor masks 1,2,4,8
    float mnew[8], rs[8];
#pragma unroll
    for (int i = 0; i < 8; ++i) {
      float tm = fmaxf(fmaxf(sc[0][i], sc[1][i]), fmaxf(sc[2][i], sc[3][i]));
#pragma unroll
      for (int x = 1; x < 16; x <<= 1) tm = fmaxf(tm, __shfl_xor(tm, x, 32));
      mnew[i] = fmaxf(mrow[i], tm);
      float ps = 0.0f;
#pragma unroll
      for (int nt = 0; nt < 4; ++nt) {
        float p = __expf(sc[nt][i] - mnew[i]);
        sc[nt][i] = p;
        ps += p;
      }
#pragma unroll
      for (int x = 1; x < 16; x <<= 1) ps += __shfl_xor(ps, x, 32);
      rs[i] = ps;
    }
#pragma unroll
    for (int i = 0; i < 8; ++i) {
      float f = __expf(mrow[i] - mnew[i]);
      lrow[i] = lrow[i] * f + rs[i];
      mrow[i] = mnew[i];
#pragma unroll
      for (int nt = 0; nt < 4; ++nt) acc[nt][i] *= f;
    }

    // C-layout -> A-layout via per-wave LDS round trip
#pragma unroll
    for (int nt = 0; nt < 4; ++nt)
#pragma unroll
      for (int i = 0; i < 8; ++i)
        Pl[wave][rbase + i][nt * 16 + col] = (_Float16)sc[nt][i];
    __syncthreads();

    // O += P @ V : B-frag = contiguous V^T rows (fixed d, contiguous kv)
#pragma unroll
    for (int s = 0; s < 2; ++s) {
      v16h ap = frag_a(&Pl[wave][col][0], kbA + s * 32);
#pragma unroll
      for (int nt = 0; nt < 4; ++nt)
        acc[nt] = wmma16(ap, frag_b(&Vs[nt * 16 + col][kbB + s * 32]), acc[nt]);
    }
  }

  // epilogue: normalize by row sum, write [s][h*64+d] f16
#pragma unroll
  for (int nt = 0; nt < 4; ++nt)
#pragma unroll
    for (int i = 0; i < 8; ++i) {
      int m = q0 + wave * 16 + rbase + i;
      float val = acc[nt][i] / lrow[i];
      attn[(size_t)m * D_MODEL + head * HEAD_DIM + nt * 16 + col] =
          (_Float16)val;
    }
}

// ---------------------------------------------------------------------------
// Output projection: out(f32) = attn(f16) @ Wo + bo
// Block tile M=128, N=128, Kstep=32.
// ---------------------------------------------------------------------------
__global__ __launch_bounds__(256) void out_gemm(
    const _Float16* __restrict__ A, const _Float16* __restrict__ WT,
    const float* __restrict__ bias, float* __restrict__ out) {
  __shared__ _Float16 As[128][32];
  __shared__ _Float16 Ws[128][32];
  const int t = threadIdx.x;
  const int lane = t & 31, wave = t >> 5;
  const int m0 = blockIdx.x * 128;
  const int n0 = blockIdx.y * 128;
  const int col = lane & 15;
  const int rbase = (lane >> 4) * 8;
  const int kbA = (lane >> 4) * 8;
  const int kbB = (lane >> 4) * 8;

  v8f acc[8] = {};

  for (int k0 = 0; k0 < D_MODEL; k0 += 32) {
    __syncthreads();
#pragma unroll
    for (int it = 0; it < 2; ++it) {  // 128x32 f16: 512 groups of 8
      int g = t + it * 256;
      int row = g >> 2;
      int cg = (g & 3) * 8;
      *(v8h*)&As[row][cg] =
          *(const v8h*)(A + (size_t)(m0 + row) * D_MODEL + k0 + cg);
    }
#pragma unroll
    for (int it = 0; it < 2; ++it) {
      int g = t + it * 256;
      int row = g >> 2;
      int cg = (g & 3) * 8;
      *(v8h*)&Ws[row][cg] =
          *(const v8h*)(WT + (size_t)(n0 + row) * D_MODEL + k0 + cg);
    }
    __syncthreads();

    v16h a = frag_a(&As[wave * 16 + col][0], kbA);
#pragma unroll
    for (int nt = 0; nt < 8; ++nt)
      acc[nt] = wmma16(a, frag_b(&Ws[nt * 16 + col][kbB]), acc[nt]);
  }

#pragma unroll
  for (int nt = 0; nt < 8; ++nt) {
    int n = n0 + nt * 16 + col;
    float bv = bias[n];
#pragma unroll
    for (int i = 0; i < 8; ++i) {
      int m = m0 + wave * 16 + rbase + i;
      out[(size_t)m * D_MODEL + n] = acc[nt][i] + bv;
    }
  }
}

// ---------------------------------------------------------------------------
extern "C" void kernel_launch(void* const* d_in, const int* in_sizes, int n_in,
                              void* d_out, int out_size, void* d_ws,
                              size_t ws_size, hipStream_t stream) {
  const float* query = (const float*)d_in[0];
  const float* key   = (const float*)d_in[1];
  const float* value = (const float*)d_in[2];
  const int*   amask = (const int*)d_in[3];
  const float* Wq = (const float*)d_in[4];
  const float* bq = (const float*)d_in[5];
  const float* Wk = (const float*)d_in[6];
  const float* bk = (const float*)d_in[7];
  const float* Wv = (const float*)d_in[8];
  const float* bv = (const float*)d_in[9];
  const float* Wo = (const float*)d_in[10];
  const float* bo = (const float*)d_in[11];
  float* out = (float*)d_out;

  char* ws = (char*)d_ws;
  const size_t MATB = (size_t)D_MODEL * D_MODEL * sizeof(_Float16);  // 2 MB
  const size_t ACTB = (size_t)S_LEN * D_MODEL * sizeof(_Float16);    // 8 MB
  _Float16* WqT  = (_Float16*)(ws + 0 * MATB);
  _Float16* WkT  = (_Float16*)(ws + 1 * MATB);
  _Float16* WvT  = (_Float16*)(ws + 2 * MATB);
  _Float16* WoT  = (_Float16*)(ws + 3 * MATB);
  _Float16* qbuf = (_Float16*)(ws + 4 * MATB);
  _Float16* kbuf = (_Float16*)(ws + 4 * MATB + 1 * ACTB);
  _Float16* vtbf = (_Float16*)(ws + 4 * MATB + 2 * ACTB);
  _Float16* abuf = (_Float16*)(ws + 4 * MATB + 3 * ACTB);

  dim3 tb(256);
  dim3 gtr(D_MODEL / 32, D_MODEL / 32);
  wtrans_f16<<<gtr, tb, 0, stream>>>(Wq, WqT);
  wtrans_f16<<<gtr, tb, 0, stream>>>(Wk, WkT);
  wtrans_f16<<<gtr, tb, 0, stream>>>(Wv, WvT);
  wtrans_f16<<<gtr, tb, 0, stream>>>(Wo, WoT);

  dim3 gg(S_LEN / 128, D_MODEL / 128);
  proj_gemm<0><<<gg, tb, 0, stream>>>(query, WqT, bq, qbuf, 0.125f);  // 1/sqrt(64)
  proj_gemm<0><<<gg, tb, 0, stream>>>(key,   WkT, bk, kbuf, 1.0f);
  proj_gemm<1><<<gg, tb, 0, stream>>>(value, WvT, bv, vtbf, 1.0f);

  flash_attn<<<dim3(S_LEN / 128, N_HEADS), tb, 0, stream>>>(qbuf, kbuf, vtbf,
                                                            amask, abuf);

  out_gemm<<<gg, tb, 0, stream>>>(abuf, WoT, bo, out);
}
